// ImprovedTeacherStudentLoss_16131897164008
// MI455X (gfx1250) — matrix operand order repaired and checked by reading, not compiled, not run on hardware
//
#include <hip/hip_runtime.h>
#include <math.h>

// ---------------------------------------------------------------------------
// ImprovedTeacherStudentLoss for MI455X (gfx1250, wave32, WMMA).
// Dominant cost: 8192x8192x1024 GEMM (137 GFLOP) fused with online masked
// logsumexp. Split-precision bf16x3 (hi/lo decomposition, 3x
// V_WMMA_F32_16X16X32_BF16 per K=32) => ~fp32 accuracy at bf16-matrix rate.
// Each wave owns 2 M-tiles x 8 N-tiles (16 accumulators) to amortize B
// fragment LDS traffic: 40 ds_load_b128 per 48 WMMAs (0.83 loads/WMMA).
// N-dimension split 8 ways for occupancy with deterministic (m,s,pos) merge.
// ---------------------------------------------------------------------------

typedef __attribute__((ext_vector_type(8)))  float  v8f;
typedef __attribute__((ext_vector_type(8)))  __bf16 v8bf;
typedef __attribute__((ext_vector_type(16))) __bf16 v16bf;

#define TEMP    0.07f
#define NEG_BIG (-1.0e30f)
#define NSPLIT  8
#define ROWW    40   // padded LDS row: 40 bf16 = 80 B = 20 words (4*odd) ->
                     // conflict-free b128 frag loads, 16B-aligned stores

// ---------------------------------------------------------------------------
// Kernel A: classification CE + distillation KL over [B, 2] logits.
// ---------------------------------------------------------------------------
__global__ __launch_bounds__(256) void logits_losses_kernel(
    const float* __restrict__ slog, const float* __restrict__ tlog,
    const int* __restrict__ labels, float* __restrict__ scal, int B)
{
    const int tid = threadIdx.x;
    const float invT = 1.0f / TEMP;
    float cls = 0.0f, dist = 0.0f;
    for (int r = tid; r < B; r += 256) {
        float s0 = slog[2 * r], s1 = slog[2 * r + 1];
        float m  = fmaxf(s0, s1);
        float lse = m + logf(expf(s0 - m) + expf(s1 - m));
        int lab = labels[r];
        cls += lse - (lab ? s1 : s0);

        float ts0 = s0 * invT, ts1 = s1 * invT;
        float t0 = tlog[2 * r] * invT, t1 = tlog[2 * r + 1] * invT;
        float tm = fmaxf(t0, t1);
        float tlse = tm + logf(expf(t0 - tm) + expf(t1 - tm));
        float sm = fmaxf(ts0, ts1);
        float slse = sm + logf(expf(ts0 - sm) + expf(ts1 - sm));
        float tlp0 = t0 - tlse, tlp1 = t1 - tlse;
        float slp0 = ts0 - slse, slp1 = ts1 - slse;
        dist += expf(tlp0) * (tlp0 - slp0) + expf(tlp1) * (tlp1 - slp1);
    }
    __shared__ float red0[256];
    __shared__ float red1[256];
    red0[tid] = cls; red1[tid] = dist;
    __syncthreads();
    for (int s = 128; s > 0; s >>= 1) {
        if (tid < s) { red0[tid] += red0[tid + s]; red1[tid] += red1[tid + s]; }
        __syncthreads();
    }
    if (tid == 0) {
        scal[0] = red0[0] / (float)B;                     // cls_loss
        scal[1] = red1[0] / (float)B * (TEMP * TEMP);     // distill_loss
    }
}

// ---------------------------------------------------------------------------
// Kernel B: per-row L2 norms, cosine(student, teacher), and hi/lo bf16
// decomposition of f = [norm(student); norm(aug)]. One block per row.
// ---------------------------------------------------------------------------
__global__ __launch_bounds__(256) void normalize_cos_kernel(
    const float* __restrict__ sf, const float* __restrict__ tf,
    const float* __restrict__ af,
    __bf16* __restrict__ f_hi, __bf16* __restrict__ f_lo,
    float* __restrict__ cos_arr, int B, int D)
{
    const int r   = blockIdx.x;
    const int tid = threadIdx.x;
    const float* s = sf + (size_t)r * D;
    const float* t = tf + (size_t)r * D;
    const float* a = af + (size_t)r * D;

    float ss = 0.f, st = 0.f, sa = 0.f, dp = 0.f;
    for (int i = tid; i < D; i += 256) {
        float vs = s[i], vt = t[i], va = a[i];
        ss += vs * vs; st += vt * vt; sa += va * va; dp += vs * vt;
    }
    __shared__ float red[4][256];
    red[0][tid] = ss; red[1][tid] = st; red[2][tid] = sa; red[3][tid] = dp;
    __syncthreads();
    for (int k = 128; k > 0; k >>= 1) {
        if (tid < k) {
            red[0][tid] += red[0][tid + k];
            red[1][tid] += red[1][tid + k];
            red[2][tid] += red[2][tid + k];
            red[3][tid] += red[3][tid + k];
        }
        __syncthreads();
    }
    const float ns = fmaxf(sqrtf(red[0][0]), 1e-12f);
    const float nt = fmaxf(sqrtf(red[1][0]), 1e-12f);
    const float na = fmaxf(sqrtf(red[2][0]), 1e-12f);
    if (tid == 0) cos_arr[r] = red[3][0] / (ns * nt);

    const float inv_s = 1.0f / ns, inv_a = 1.0f / na;
    __bf16* hs = f_hi + (size_t)r * D;
    __bf16* ls = f_lo + (size_t)r * D;
    __bf16* ha = f_hi + (size_t)(B + r) * D;
    __bf16* la = f_lo + (size_t)(B + r) * D;
    for (int i = tid; i < D; i += 256) {
        float xs = s[i] * inv_s;
        __bf16 h = (__bf16)xs;
        hs[i] = h;
        ls[i] = (__bf16)(xs - (float)h);
        float xa = a[i] * inv_a;
        __bf16 h2 = (__bf16)xa;
        ha[i] = h2;
        la[i] = (__bf16)(xa - (float)h2);
    }
}

// ---------------------------------------------------------------------------
// bf16 WMMA fragment loaders (wave32 layouts per CDNA5 ISA 7.12.2).
// A 16x32: lane(m=l&15, h=l>>4) holds K = 8h+0..7 and 16+8h+0..7.
// B 32x16: lane(n=l&15, h=l>>4) holds K = 8h+0..7 and 16+8h+0..7.
// ---------------------------------------------------------------------------
__device__ __forceinline__ v16bf load_frag_a(const __bf16* rowbase, int h)
{
    v8bf r0 = *reinterpret_cast<const v8bf*>(rowbase + 8 * h);
    v8bf r1 = *reinterpret_cast<const v8bf*>(rowbase + 16 + 8 * h);
    return __builtin_shufflevector(r0, r1, 0, 1, 2, 3, 4, 5, 6, 7,
                                   8, 9, 10, 11, 12, 13, 14, 15);
}
__device__ __forceinline__ v16bf load_frag_b(const __bf16* rowbase, int h)
{
    v8bf r0 = *reinterpret_cast<const v8bf*>(rowbase + 8 * h);
    v8bf r1 = *reinterpret_cast<const v8bf*>(rowbase + 16 + 8 * h);
    return __builtin_shufflevector(r0, r1, 0, 1, 2, 3, 4, 5, 6, 7,
                                   8, 9, 10, 11, 12, 13, 14, 15);
}

// ---------------------------------------------------------------------------
// Kernel C: fused  sim = (f f^T)/T  +  streaming masked logsumexp + pos pick.
// Block = 128 threads (4 waves), 128 rows per block; wave owns 32 rows
// (2 M-tiles) and computes 2x8 16x16 tiles per 128-col N-block via bf16x3
// WMMA. Grid.y splits the N range NSPLIT ways; per-(row,split) partial
// (m, s, pos) merged deterministically in finalize.
// ---------------------------------------------------------------------------
__global__ __launch_bounds__(128) void simclr_gemm_lse_kernel(
    const __bf16* __restrict__ f_hi, const __bf16* __restrict__ f_lo,
    float* __restrict__ part_m, float* __restrict__ part_s,
    float* __restrict__ part_p, int N2, int D, int B)
{
    __shared__ __bf16 lds_ah[128 * ROWW];
    __shared__ __bf16 lds_al[128 * ROWW];
    __shared__ __bf16 lds_bh[128 * ROWW];
    __shared__ __bf16 lds_bl[128 * ROWW];

    const int tid   = threadIdx.x;
    const int wid   = tid >> 5;        // wave 0..3
    const int lane  = tid & 31;
    const int lhalf = lane >> 4;       // 0 / 1
    const int l15   = lane & 15;
    const int m0    = blockIdx.x * 128;
    const int split = blockIdx.y;
    const int nspan = N2 / NSPLIT;
    const int n_begin = split * nspan;
    const int n_end   = n_begin + nspan;
    const int wrow0 = m0 + wid * 32;   // wave owns rows wrow0 .. wrow0+31
    const float invT = 1.0f / TEMP;

    float run_m[16], run_s[16], posv[16];   // [mt*8 + v]
#pragma unroll
    for (int v = 0; v < 16; ++v) { run_m[v] = NEG_BIG; run_s[v] = 0.f; posv[v] = NEG_BIG; }

    for (int n0 = n_begin; n0 < n_end; n0 += 128) {
        v8f acc[16];                        // [mt*8 + nt]
#pragma unroll
        for (int i = 0; i < 16; ++i) acc[i] = v8f{};

        for (int k0 = 0; k0 < D; k0 += 32) {
            // Stage A/B tiles (128 rows x 32 bf16, hi+lo each): 512 uint4 per
            // array, 4 per thread per array.
#pragma unroll
            for (int i = 0; i < 4; ++i) {
                int idx = tid + i * 128;        // 0..511
                int rr  = idx >> 2;             // 0..127
                int c8  = (idx & 3) * 8;        // 0,8,16,24
                size_t ga = (size_t)(m0 + rr) * D + k0 + c8;
                size_t gb = (size_t)(n0 + rr) * D + k0 + c8;
                *reinterpret_cast<uint4*>(&lds_ah[rr * ROWW + c8]) =
                    *reinterpret_cast<const uint4*>(f_hi + ga);
                *reinterpret_cast<uint4*>(&lds_al[rr * ROWW + c8]) =
                    *reinterpret_cast<const uint4*>(f_lo + ga);
                *reinterpret_cast<uint4*>(&lds_bh[rr * ROWW + c8]) =
                    *reinterpret_cast<const uint4*>(f_hi + gb);
                *reinterpret_cast<uint4*>(&lds_bl[rr * ROWW + c8]) =
                    *reinterpret_cast<const uint4*>(f_lo + gb);
            }
            __syncthreads();

            // A fragments for both M-tiles (reused across all 8 N-tiles).
            v16bf ah0 = load_frag_a(&lds_ah[(wid * 32 + l15) * ROWW], lhalf);
            v16bf al0 = load_frag_a(&lds_al[(wid * 32 + l15) * ROWW], lhalf);
            v16bf ah1 = load_frag_a(&lds_ah[(wid * 32 + 16 + l15) * ROWW], lhalf);
            v16bf al1 = load_frag_a(&lds_al[(wid * 32 + 16 + l15) * ROWW], lhalf);
#pragma unroll
            for (int nt = 0; nt < 8; ++nt) {
                v16bf bh = load_frag_b(&lds_bh[(nt * 16 + l15) * ROWW], lhalf);
                v16bf bl = load_frag_b(&lds_bl[(nt * 16 + l15) * ROWW], lhalf);
                // hi*hi + hi*lo + lo*hi  (lo*lo ~2^-18, dropped)
                acc[nt] = __builtin_amdgcn_wmma_f32_16x16x32_bf16(
                    false, ah0, false, bh, (short)0, acc[nt], false, false);
                acc[nt] = __builtin_amdgcn_wmma_f32_16x16x32_bf16(
                    false, ah0, false, bl, (short)0, acc[nt], false, false);
                acc[nt] = __builtin_amdgcn_wmma_f32_16x16x32_bf16(
                    false, al0, false, bh, (short)0, acc[nt], false, false);
                acc[8 + nt] = __builtin_amdgcn_wmma_f32_16x16x32_bf16(
                    false, ah1, false, bh, (short)0, acc[8 + nt], false, false);
                acc[8 + nt] = __builtin_amdgcn_wmma_f32_16x16x32_bf16(
                    false, ah1, false, bl, (short)0, acc[8 + nt], false, false);
                acc[8 + nt] = __builtin_amdgcn_wmma_f32_16x16x32_bf16(
                    false, al1, false, bh, (short)0, acc[8 + nt], false, false);
            }
            __syncthreads();
        }

        // Epilogue: scale, mask diagonal, capture positive, online LSE update.
#pragma unroll
        for (int mt = 0; mt < 2; ++mt) {
#pragma unroll
            for (int v = 0; v < 8; ++v) {
                const int r    = wrow0 + mt * 16 + v + 8 * lhalf;
                const int pcol = (r + B) % N2;
                const int ri   = mt * 8 + v;
                float x[8];
                float tm = NEG_BIG;
#pragma unroll
                for (int nt = 0; nt < 8; ++nt) {
                    const int j = n0 + nt * 16 + l15;
                    float xv = acc[mt * 8 + nt][v] * invT;
                    if (j == r) xv = NEG_BIG;           // drop self-sim
                    if (j == pcol) posv[ri] = fmaxf(posv[ri], xv);
                    x[nt] = xv;
                    tm = fmaxf(tm, xv);
                }
                for (int msk = 1; msk <= 8; msk <<= 1)
                    tm = fmaxf(tm, __shfl_xor(tm, msk, 32));
                const float nm = fmaxf(run_m[ri], tm);
                float ssum = 0.f;
#pragma unroll
                for (int nt = 0; nt < 8; ++nt) ssum += expf(x[nt] - nm);
                for (int msk = 1; msk <= 8; msk <<= 1)
                    ssum += __shfl_xor(ssum, msk, 32);
                run_s[ri] = run_s[ri] * expf(run_m[ri] - nm) + ssum;
                run_m[ri] = nm;
            }
        }
    }

    // Write per-(row, split) partials.
#pragma unroll
    for (int mt = 0; mt < 2; ++mt) {
#pragma unroll
        for (int v = 0; v < 8; ++v) {
            const int ri = mt * 8 + v;
            float p = posv[ri];
            for (int msk = 1; msk <= 8; msk <<= 1)
                p = fmaxf(p, __shfl_xor(p, msk, 32));
            if (l15 == 0) {
                const int r = wrow0 + mt * 16 + v + 8 * lhalf;
                part_m[(size_t)split * N2 + r] = run_m[ri];
                part_s[(size_t)split * N2 + r] = run_s[ri];
                part_p[(size_t)split * N2 + r] = p;
            }
        }
    }
}

// ---------------------------------------------------------------------------
// Kernel D: merge N-split partials, final deterministic reductions, outputs.
// ---------------------------------------------------------------------------
__global__ __launch_bounds__(256) void finalize_kernel(
    const float* __restrict__ cos_arr,
    const float* __restrict__ part_m, const float* __restrict__ part_s,
    const float* __restrict__ part_p,
    const float* __restrict__ scal, float* __restrict__ out, int B, int N2)
{
    const int tid = threadIdx.x;
    float c = 0.f, s = 0.f;
    for (int i = tid; i < B; i += 256) c += cos_arr[i];
    for (int r = tid; r < N2; r += 256) {
        float m = part_m[r];
#pragma unroll
        for (int sp = 1; sp < NSPLIT; ++sp)
            m = fmaxf(m, part_m[(size_t)sp * N2 + r]);
        float se = 0.f, p = NEG_BIG;
#pragma unroll
        for (int sp = 0; sp < NSPLIT; ++sp) {
            se += part_s[(size_t)sp * N2 + r] * expf(part_m[(size_t)sp * N2 + r] - m);
            p = fmaxf(p, part_p[(size_t)sp * N2 + r]);
        }
        s += (m + logf(se)) - p;   // lse - pos for this row
    }
    __shared__ float red0[256];
    __shared__ float red1[256];
    red0[tid] = c; red1[tid] = s;
    __syncthreads();
    for (int k = 128; k > 0; k >>= 1) {
        if (tid < k) { red0[tid] += red0[tid + k]; red1[tid] += red1[tid + k]; }
        __syncthreads();
    }
    if (tid == 0) {
        const float feature = 1.0f - red0[0] / (float)B;
        const float simclr  = red1[0] / (float)N2;
        const float cls     = scal[0];
        const float dist    = scal[1];
        out[0] = 1.0f * cls + 1.0f * dist + 0.5f * feature + 0.3f * simclr;
        out[1] = cls;
        out[2] = dist;
        out[3] = feature;
        out[4] = simclr;
    }
}

// ---------------------------------------------------------------------------
extern "C" void kernel_launch(void* const* d_in, const int* in_sizes, int n_in,
                              void* d_out, int out_size, void* d_ws, size_t ws_size,
                              hipStream_t stream)
{
    const float* sf    = (const float*)d_in[0];  // student_features   [B, D]
    const float* slog  = (const float*)d_in[1];  // student_logits     [B, 2]
    const float* tf    = (const float*)d_in[2];  // teacher_features   [B, D]
    const float* tlog  = (const float*)d_in[3];  // teacher_logits     [B, 2]
    const int*   labs  = (const int*)  d_in[4];  // labels             [B]
    const float* af    = (const float*)d_in[5];  // student_features_aug [B, D]

    const int B  = in_sizes[4];
    const int D  = in_sizes[0] / B;
    const int N2 = 2 * B;

    // Workspace layout (bf16 hi/lo = 4 B/elem total, same as fp32 f was).
    char* ws = (char*)d_ws;
    __bf16* f_hi  = (__bf16*)ws;                            // N2*D bf16
    __bf16* f_lo  = f_hi + (size_t)N2 * D;                  // N2*D bf16
    float* cos_arr = (float*)(f_lo + (size_t)N2 * D);       // B
    float* part_m  = cos_arr + B;                           // NSPLIT*N2
    float* part_s  = part_m + (size_t)NSPLIT * N2;          // NSPLIT*N2
    float* part_p  = part_s + (size_t)NSPLIT * N2;          // NSPLIT*N2
    float* scal    = part_p + (size_t)NSPLIT * N2;          // 2

    logits_losses_kernel<<<1, 256, 0, stream>>>(slog, tlog, labs, scal, B);
    normalize_cos_kernel<<<B, 256, 0, stream>>>(sf, tf, af, f_hi, f_lo, cos_arr, B, D);
    dim3 grid(N2 / 128, NSPLIT);
    simclr_gemm_lse_kernel<<<grid, 128, 0, stream>>>(
        f_hi, f_lo, part_m, part_s, part_p, N2, D, B);
    finalize_kernel<<<1, 256, 0, stream>>>(
        cos_arr, part_m, part_s, part_p, scal, (float*)d_out, B, N2);
}
